// SIMNet_88673894793292
// MI455X (gfx1250) — matrix operand order repaired and checked by reading, not compiled, not run on hardware
//
#include <hip/hip_runtime.h>

typedef __attribute__((ext_vector_type(16))) _Float16 v16h;
typedef __attribute__((ext_vector_type(8)))  _Float16 v8h;
typedef __attribute__((ext_vector_type(4)))  _Float16 v4h;
typedef __attribute__((ext_vector_type(8)))  float    v8f;
typedef __attribute__((ext_vector_type(4)))  float    v4f;

#define N_NODES 16384
#define DIMK    1024
#define EPS_F   1e-8f

// ---------------------------------------------------------------- min/max(Age)
__global__ void age_minmax_kernel(const float* __restrict__ Age, int n,
                                  float* __restrict__ mm) {
    __shared__ float smin[256];
    __shared__ float smax[256];
    int t = threadIdx.x;
    float mn = 3.4e38f, mx = -3.4e38f;
    for (int i = t; i < n; i += 256) {
        float a = Age[i];
        mn = fminf(mn, a);
        mx = fmaxf(mx, a);
    }
    smin[t] = mn; smax[t] = mx;
    __syncthreads();
    for (int s = 128; s > 0; s >>= 1) {
        if (t < s) {
            smin[t] = fminf(smin[t], smin[t + s]);
            smax[t] = fmaxf(smax[t], smax[t + s]);
        }
        __syncthreads();
    }
    if (t == 0) { mm[0] = smin[0]; mm[1] = smax[0]; }
}

// ---------------------------------------------------------------- f32 -> f16
__global__ void f32_to_f16_kernel(const float* __restrict__ in,
                                  _Float16* __restrict__ out, int n) {
    int i = (blockIdx.x * blockDim.x + threadIdx.x) * 4;
    if (i < n) {
        v4f v = *(const v4f*)(in + i);
        v4h h;
        h.x = (_Float16)v.x; h.y = (_Float16)v.y;
        h.z = (_Float16)v.z; h.w = (_Float16)v.w;
        *(v4h*)(out + i) = h;
    }
}

// ------------------------------------------- h = sigmoid(feat @ W^T + b), WMMA
// A = featH [16384][1024] f16 row-major
// B rows   = WH   [1024][1024] f16 row-major; B[k][n] = W[n][k] -> contiguous K
// Block: 128x128 tile, 8 waves (4 in M x 2 in N); wave tile 32x64 (2x4 WMMAs)
__global__ __launch_bounds__(256, 1)
void gemm_sigmoid_kernel(const _Float16* __restrict__ A,
                         const _Float16* __restrict__ W,
                         const float* __restrict__ bias,
                         float* __restrict__ H) {
    const int lane = threadIdx.x & 31;
    const int wave = threadIdx.x >> 5;
    const int wm = wave & 3;      // 0..3  -> M offset wm*32
    const int wn = wave >> 2;     // 0..1  -> N offset wn*64
    const int m0 = blockIdx.x * 128 + wm * 32;
    const int n0 = blockIdx.y * 128 + wn * 64;

    // A fragment addressing (16-bit A 16x32 layout, contiguous K to match B):
    //   lane<16 : row m0+lane,    K {0..15}
    //   lane>=16: row m0+lane-16, K {16..31}
    const int arow  = lane & 15;
    const int akoff = (lane >> 4) << 4;   // 0 or 16
    // B fragment addressing (16-bit B 32x16 layout):
    //   lane<16 : col n, K {0..15} ; lane>=16: col n, K {16..31} (contiguous)
    const int bcol  = lane & 15;
    const int bkoff = (lane >> 4) << 4;   // 0 or 16

    v8f zero;
    #pragma unroll
    for (int q = 0; q < 8; ++q) zero[q] = 0.0f;
    v8f acc[2][4];
    #pragma unroll
    for (int i = 0; i < 2; ++i)
        #pragma unroll
        for (int j = 0; j < 4; ++j) acc[i][j] = zero;

    const _Float16* aptr0 = A + (size_t)(m0 + arow) * DIMK + akoff;
    const _Float16* aptr1 = aptr0 + (size_t)16 * DIMK;
    const _Float16* bptr0 = W + (size_t)(n0 +  0 + bcol) * DIMK + bkoff;
    const _Float16* bptr1 = W + (size_t)(n0 + 16 + bcol) * DIMK + bkoff;
    const _Float16* bptr2 = W + (size_t)(n0 + 32 + bcol) * DIMK + bkoff;
    const _Float16* bptr3 = W + (size_t)(n0 + 48 + bcol) * DIMK + bkoff;

    union Frag { v16h v; v8h h[2]; };

    for (int k0 = 0; k0 < DIMK; k0 += 32) {
        // speculative prefetch of A a few K-steps ahead (global_prefetch_b8)
        __builtin_prefetch((const void*)(aptr0 + k0 + 256), 0, 0);
        __builtin_prefetch((const void*)(aptr1 + k0 + 256), 0, 0);

        Frag a0, a1;
        a0.h[0] = *(const v8h*)(aptr0 + k0);
        a0.h[1] = *(const v8h*)(aptr0 + k0 + 8);
        a1.h[0] = *(const v8h*)(aptr1 + k0);
        a1.h[1] = *(const v8h*)(aptr1 + k0 + 8);

        Frag b0, b1, b2, b3;
        b0.h[0] = *(const v8h*)(bptr0 + k0);  b0.h[1] = *(const v8h*)(bptr0 + k0 + 8);
        b1.h[0] = *(const v8h*)(bptr1 + k0);  b1.h[1] = *(const v8h*)(bptr1 + k0 + 8);
        b2.h[0] = *(const v8h*)(bptr2 + k0);  b2.h[1] = *(const v8h*)(bptr2 + k0 + 8);
        b3.h[0] = *(const v8h*)(bptr3 + k0);  b3.h[1] = *(const v8h*)(bptr3 + k0 + 8);

        acc[0][0] = __builtin_amdgcn_wmma_f32_16x16x32_f16(false, a0.v, false, b0.v, (short)0, acc[0][0], false, false);
        acc[1][0] = __builtin_amdgcn_wmma_f32_16x16x32_f16(false, a1.v, false, b0.v, (short)0, acc[1][0], false, false);
        acc[0][1] = __builtin_amdgcn_wmma_f32_16x16x32_f16(false, a0.v, false, b1.v, (short)0, acc[0][1], false, false);
        acc[1][1] = __builtin_amdgcn_wmma_f32_16x16x32_f16(false, a1.v, false, b1.v, (short)0, acc[1][1], false, false);
        acc[0][2] = __builtin_amdgcn_wmma_f32_16x16x32_f16(false, a0.v, false, b2.v, (short)0, acc[0][2], false, false);
        acc[1][2] = __builtin_amdgcn_wmma_f32_16x16x32_f16(false, a1.v, false, b2.v, (short)0, acc[1][2], false, false);
        acc[0][3] = __builtin_amdgcn_wmma_f32_16x16x32_f16(false, a0.v, false, b3.v, (short)0, acc[0][3], false, false);
        acc[1][3] = __builtin_amdgcn_wmma_f32_16x16x32_f16(false, a1.v, false, b3.v, (short)0, acc[1][3], false, false);
    }

    // C/D layout: VGPR r -> row = r + 8*(lane>=16), col = lane&15
    const int crow = (lane >> 4) << 3;
    const int ccol = lane & 15;
    #pragma unroll
    for (int j = 0; j < 4; ++j) {
        const int col = n0 + j * 16 + ccol;
        const float bv = bias[col];
        #pragma unroll
        for (int i = 0; i < 2; ++i) {
            #pragma unroll
            for (int r = 0; r < 8; ++r) {
                const int row = m0 + i * 16 + crow + r;
                const float x = acc[i][j][r] + bv;
                H[(size_t)row * DIMK + col] = 1.0f / (1.0f + __expf(-x));
            }
        }
    }
}

// ------------------------------- per-node: age=tanh(minmax(Age)), inv norms
__global__ void node_stats_kernel(const float* __restrict__ H,
                                  const float* __restrict__ Age,
                                  const float* __restrict__ Sex,
                                  const float* __restrict__ mm,
                                  float* __restrict__ age_out,
                                  float* __restrict__ invn_out, int n) {
    const int lane = threadIdx.x & 31;
    const int node = blockIdx.x * 8 + (threadIdx.x >> 5);
    if (node >= n) return;
    const float* row = H + (size_t)node * DIMK;
    float s = 0.0f;
    #pragma unroll
    for (int c = 0; c < 8; ++c) {
        v4f v = *(const v4f*)(row + c * 128 + lane * 4);
        s += v.x * v.x + v.y * v.y + v.z * v.z + v.w * v.w;
    }
    #pragma unroll
    for (int m = 16; m > 0; m >>= 1) s += __shfl_xor(s, m, 32);
    const float mn = mm[0], mx = mm[1];
    const float ag = tanhf((Age[node] - mn) / (mx - mn));
    const float sx = Sex[node];
    const float nrm = sqrtf(s + ag * ag + sx * sx);
    const float inv = 1.0f / fmaxf(nrm, EPS_F);
    if (lane == 0) { age_out[node] = ag; invn_out[node] = inv; }
}

// ---------------------------------------------------- per-edge cosine sim
__global__ void edge_sim_kernel(const float* __restrict__ H,
                                const int* __restrict__ src,
                                const int* __restrict__ dst,
                                const float* __restrict__ age,
                                const float* __restrict__ sex,
                                const float* __restrict__ invn,
                                float* __restrict__ out, int nE) {
    const int lane = threadIdx.x & 31;
    const int e = blockIdx.x * 8 + (threadIdx.x >> 5);
    if (e >= nE) return;
    const int s = src[e], d = dst[e];
    const float* rs = H + (size_t)s * DIMK;
    const float* rd = H + (size_t)d * DIMK;
    float acc = 0.0f;
    #pragma unroll
    for (int c = 0; c < 8; ++c) {
        v4f a = *(const v4f*)(rs + c * 128 + lane * 4);
        v4f b = *(const v4f*)(rd + c * 128 + lane * 4);
        acc += a.x * b.x + a.y * b.y + a.z * b.z + a.w * b.w;
    }
    #pragma unroll
    for (int m = 16; m > 0; m >>= 1) acc += __shfl_xor(acc, m, 32);
    if (lane == 0) {
        acc += age[s] * age[d] + sex[s] * sex[d];
        out[e] = acc * invn[s] * invn[d];
    }
}

// ---------------------------------------------------------------- launcher
extern "C" void kernel_launch(void* const* d_in, const int* in_sizes, int n_in,
                              void* d_out, int out_size, void* d_ws, size_t ws_size,
                              hipStream_t stream) {
    const float* feat = (const float*)d_in[0];   // [N, D]
    const float* Age  = (const float*)d_in[1];   // [N]
    const float* Sex  = (const float*)d_in[2];   // [N]
    const float* W    = (const float*)d_in[3];   // [D, D]
    const float* bias = (const float*)d_in[4];   // [D]
    const int*   src  = (const int*)d_in[5];     // [E]
    const int*   dst  = (const int*)d_in[6];     // [E]
    float* out = (float*)d_out;                  // [E] f32

    const int N = in_sizes[1];
    const int E = in_sizes[5];

    // workspace layout (256B aligned)
    char* ws = (char*)d_ws;
    size_t off = 0;
    _Float16* featH = (_Float16*)(ws + off); off += (size_t)N_NODES * DIMK * 2;   // 33.5 MB
    _Float16* WH    = (_Float16*)(ws + off); off += (size_t)DIMK * DIMK * 2;      //  2.1 MB
    float*    H     = (float*)(ws + off);    off += (size_t)N_NODES * DIMK * 4;   // 67.1 MB
    float*    mm    = (float*)(ws + off);    off += 256;
    float*    ageA  = (float*)(ws + off);    off += (size_t)N_NODES * 4;
    float*    invn  = (float*)(ws + off);    off += (size_t)N_NODES * 4;
    (void)ws_size; (void)n_in; (void)out_size;

    // 1) Age min/max
    age_minmax_kernel<<<1, 256, 0, stream>>>(Age, N, mm);

    // 2) f32 -> f16 conversions (feat, W)
    {
        const int nf = N_NODES * DIMK;          // 16,777,216
        f32_to_f16_kernel<<<nf / (256 * 4), 256, 0, stream>>>(feat, featH, nf);
        const int nw = DIMK * DIMK;             // 1,048,576
        f32_to_f16_kernel<<<nw / (256 * 4), 256, 0, stream>>>(W, WH, nw);
    }

    // 3) WMMA GEMM + bias + sigmoid -> H
    {
        dim3 grid(N_NODES / 128, DIMK / 128);   // 128 x 8
        gemm_sigmoid_kernel<<<grid, 256, 0, stream>>>(featH, WH, bias, H);
    }

    // 4) per-node age + inverse norm
    node_stats_kernel<<<(N + 7) / 8, 256, 0, stream>>>(H, Age, Sex, mm, ageA, invn, N);

    // 5) per-edge cosine similarity
    edge_sim_kernel<<<(E + 7) / 8, 256, 0, stream>>>(H, src, dst, ageA, Sex, invn, out, E);
}
